// LocalLinOSS_89928025244483
// MI455X (gfx1250) — matrix operand order repaired and checked by reading, not compiled, not run on hardware
//
#include <hip/hip_runtime.h>

// ---------------------------------------------------------------------------
// LocalLinOSS on MI455X (gfx1250): diagonal-recurrence factorization.
// All heavy math is fp16 WMMA (v_wmma_f32_16x16x32_f16) with fp32 accum.
// ---------------------------------------------------------------------------

#define LL 4096
#define DD 256
#define SS 256
#define NBB 4
#define VV 8000
#define CHUNK 64
#define NCHUNK (LL / CHUNK)

typedef __attribute__((ext_vector_type(16))) _Float16 v16h;
typedef __attribute__((ext_vector_type(8)))  _Float16 v8h;
typedef __attribute__((ext_vector_type(8)))  float    v8f;

// ---------------------------------------------------------------- helpers
__device__ __forceinline__ float softplus_f(float x) {
    return (x > 20.f) ? x : log1pf(expf(x));
}

__device__ __forceinline__ float gelu_f(float x) {
    const float k = 0.7978845608028654f; // sqrt(2/pi)
    float x3 = x * x * x;
    return 0.5f * x * (1.f + tanhf(k * (x + 0.044715f * x3)));
}

// A-fragment: 16x32 fp16, M rows striped over lanes per ISA 7.12.2.
// lane<16: row=lane,    K = {k0+0..7,  k0+16..23}
// lane>=16: row=lane-16, K = {k0+8..15, k0+24..31}
__device__ __forceinline__ v16h load_a_frag(const _Float16* __restrict__ A,
                                            int K, int m, int k0, int lane) {
    int klo = (lane < 16) ? 0 : 8;
    const v8h* p0 = (const v8h*)(A + (size_t)m * K + k0 + klo);
    const v8h* p1 = (const v8h*)(A + (size_t)m * K + k0 + 16 + klo);
    union { v16h v; v8h h[2]; } u;
    u.h[0] = *p0;
    u.h[1] = *p1;
    return u.v;
}

// B-fragment: 32x16 fp16 (K x N). lane<16: col=lane, K={k0+0..7, k0+16..23};
// lane>=16: col=lane-16, K={k0+8..15, k0+24..31}. W stored [N,K] row-major, so
// each lane reads two runs of 8 contiguous halves (16B each).
__device__ __forceinline__ v16h load_b_frag(const _Float16* __restrict__ W,
                                            int K, int n, int k0, int lane) {
    int klo = (lane < 16) ? 0 : 8;
    union { v16h v; v8h h[2]; } u;
    u.h[0] = *(const v8h*)(W + (size_t)n * K + k0 + klo);
    u.h[1] = *(const v8h*)(W + (size_t)n * K + k0 + 16 + klo);
    return u.v;
}

// ---------------------------------------------------------------- GEMM
// C[M,N] = A[M,K] @ W[N,K]^T  (A fp16 row-major, W fp16 row-major [N,K])
// mode 0: Cf32 = acc
// mode 1: Cf16 = (fp16) gelu(acc + dirv[n] * extra[m*N+n])
// mode 2: Cf32 += acc + bias[n]; if Cf16 != null, Cf16 = (fp16)Cf32
// mode 3: Cf32 = acc + bias[n]
__global__ void __launch_bounds__(256)
gemm_wmma_kernel(const _Float16* __restrict__ A, const _Float16* __restrict__ W,
                 float* __restrict__ Cf32, _Float16* __restrict__ Cf16,
                 const float* __restrict__ bias, const float* __restrict__ dirv,
                 const float* __restrict__ extra, int M, int N, int K, int mode) {
    const int lane = threadIdx.x & 31;
    const int wave = threadIdx.x >> 5;
    const int row0 = blockIdx.y * 128 + (wave >> 1) * 32;
    const int col0 = blockIdx.x * 64 + (wave & 1) * 32;

    const int mA0 = row0 + (lane & 15);
    const int mA1 = mA0 + 16;
    const int nB0 = col0 + (lane & 15);
    const int nB1 = nB0 + 16;

    v8f acc[2][2];
    acc[0][0] = (v8f){};
    acc[0][1] = (v8f){};
    acc[1][0] = (v8f){};
    acc[1][1] = (v8f){};

#pragma unroll
    for (int k0 = 0; k0 < 256; k0 += 32) {  // K == 256 for every call site
        v16h a0 = load_a_frag(A, K, mA0, k0, lane);
        v16h a1 = load_a_frag(A, K, mA1, k0, lane);
        v16h b0 = load_b_frag(W, K, nB0, k0, lane);
        v16h b1 = load_b_frag(W, K, nB1, k0, lane);
        acc[0][0] = __builtin_amdgcn_wmma_f32_16x16x32_f16(false, a0, false, b0,
                                                           (short)0, acc[0][0], false, false);
        acc[0][1] = __builtin_amdgcn_wmma_f32_16x16x32_f16(false, a0, false, b1,
                                                           (short)0, acc[0][1], false, false);
        acc[1][0] = __builtin_amdgcn_wmma_f32_16x16x32_f16(false, a1, false, b0,
                                                           (short)0, acc[1][0], false, false);
        acc[1][1] = __builtin_amdgcn_wmma_f32_16x16x32_f16(false, a1, false, b1,
                                                           (short)0, acc[1][1], false, false);
    }

    // C/D layout: VGPR r holds M = r + (lane<16 ? 0 : 8), N = lane&15.
    const int mbase = (lane < 16) ? 0 : 8;
    const int ncol  = lane & 15;
#pragma unroll
    for (int tm = 0; tm < 2; ++tm) {
#pragma unroll
        for (int tn = 0; tn < 2; ++tn) {
            int colg = col0 + tn * 16 + ncol;
#pragma unroll
            for (int r = 0; r < 8; ++r) {
                int rowg = row0 + tm * 16 + mbase + r;
                float v = acc[tm][tn][r];
                size_t idx = (size_t)rowg * N + colg;
                if (mode == 0) {
                    Cf32[idx] = v;
                } else if (mode == 1) {
                    float g = gelu_f(v + dirv[colg] * extra[idx]);
                    Cf16[idx] = (_Float16)g;
                } else if (mode == 2) {
                    float nv = Cf32[idx] + v + bias[colg];
                    Cf32[idx] = nv;
                    if (Cf16) Cf16[idx] = (_Float16)nv;
                } else {
                    Cf32[idx] = v + bias[colg];
                }
            }
        }
    }
}

// ---------------------------------------------------------------- small kernels
__global__ void f32_to_f16_kernel(const float* __restrict__ src,
                                  _Float16* __restrict__ dst, int n) {
    for (int i = blockIdx.x * blockDim.x + threadIdx.x; i < n;
         i += gridDim.x * blockDim.x)
        dst[i] = (_Float16)src[i];
}

__global__ void embed_kernel(const int* __restrict__ tokens,
                             const float* __restrict__ tok_emb,
                             const float* __restrict__ pos_emb,
                             float* __restrict__ H) {
    int idx = blockIdx.x * blockDim.x + threadIdx.x;  // L*D threads
    int t = idx >> 8;
    int d = idx & 255;
    H[idx] = tok_emb[(size_t)tokens[t] * DD + d] + pos_emb[idx];
}

__global__ void coeff_kernel(const float* __restrict__ a,
                             const float* __restrict__ g,
                             const float* __restrict__ dt,
                             float* __restrict__ coeff, int n) {
    int i = blockIdx.x * blockDim.x + threadIdx.x;
    if (i < n) {
        float dte = softplus_f(dt[i]) + 1e-4f;
        coeff[i] = expf(-softplus_f(g[i]) * dte) * cosf(a[i] * dte);
    }
}

__global__ void __launch_bounds__(256)
ln_kernel(const float* __restrict__ H, const float* __restrict__ w,
          const float* __restrict__ b, float* __restrict__ XN,
          _Float16* __restrict__ XN16) {
    __shared__ float red[256];
    const int t = blockIdx.x;
    const int d = threadIdx.x;
    float x = H[(size_t)t * DD + d];
    red[d] = x;
    __syncthreads();
#pragma unroll
    for (int off = 128; off > 0; off >>= 1) {
        if (d < off) red[d] += red[d + off];
        __syncthreads();
    }
    float mean = red[0] * (1.f / DD);
    __syncthreads();
    float c = x - mean;
    red[d] = c * c;
    __syncthreads();
#pragma unroll
    for (int off = 128; off > 0; off >>= 1) {
        if (d < off) red[d] += red[d + off];
        __syncthreads();
    }
    float var = red[0] * (1.f / DD);
    float xn = c * rsqrtf(var + 1e-5f) * w[d] + b[d];
    XN[(size_t)t * DD + d] = xn;
    XN16[(size_t)t * DD + d] = (_Float16)xn;
}

// Chunked diagonal linear scan: ns[t] = c * ns[t-1] + U[t], per channel s.
__global__ void scan_pass1_kernel(const float* __restrict__ U,
                                  const float* __restrict__ coeff,
                                  float* __restrict__ Sloc,
                                  float* __restrict__ Z) {
    const int c = blockIdx.x;      // chunk
    const int s = threadIdx.x;     // channel
    const float cf = coeff[s];
    const int t0 = c * CHUNK;
    float loc = 0.f;
    for (int t = 0; t < CHUNK; ++t) {
        loc = cf * loc + U[(size_t)(t0 + t) * SS + s];
        Sloc[(size_t)(t0 + t) * SS + s] = loc;
    }
    Z[(size_t)c * SS + s] = loc;
}

__global__ void scan_pass2_kernel(const float* __restrict__ Z,
                                  const float* __restrict__ coeff,
                                  float* __restrict__ Carry) {
    const int s = threadIdx.x;
    float cf = coeff[s];
    float p2 = cf * cf, p4 = p2 * p2, p8 = p4 * p4;
    float p16 = p8 * p8, p32 = p16 * p16, cfC = p32 * p32;  // cf^64
    float car = 0.f;
    Carry[s] = 0.f;
    for (int c = 1; c < NCHUNK; ++c) {
        car = cfC * car + Z[(size_t)(c - 1) * SS + s];
        Carry[(size_t)c * SS + s] = car;
    }
}

__global__ void scan_pass3_kernel(const float* __restrict__ Sloc,
                                  const float* __restrict__ Carry,
                                  const float* __restrict__ coeff,
                                  _Float16* __restrict__ States16) {
    const int c = blockIdx.x;
    const int s = threadIdx.x;
    const float cf = coeff[s];
    const float car = Carry[(size_t)c * SS + s];
    const int t0 = c * CHUNK;
    float p = cf;  // cf^(tlocal+1)
    for (int t = 0; t < CHUNK; ++t) {
        float v = Sloc[(size_t)(t0 + t) * SS + s] + p * car;
        States16[(size_t)(t0 + t) * SS + s] = (_Float16)v;
        p *= cf;
    }
}

// ---------------------------------------------------------------- launch
extern "C" void kernel_launch(void* const* d_in, const int* in_sizes, int n_in,
                              void* d_out, int out_size, void* d_ws, size_t ws_size,
                              hipStream_t stream) {
    const int*   tokens      = (const int*)  d_in[0];
    const float* token_embed = (const float*)d_in[1];
    const float* pos_embed   = (const float*)d_in[2];
    const float* in_to_state = (const float*)d_in[3];
    const float* st_to_hid   = (const float*)d_in[4];
    const float* direct      = (const float*)d_in[5];
    const float* a_diag      = (const float*)d_in[6];
    const float* g_diag      = (const float*)d_in[7];
    const float* dt          = (const float*)d_in[8];
    const float* ln_w        = (const float*)d_in[9];
    const float* ln_b        = (const float*)d_in[10];
    const float* outp_W      = (const float*)d_in[11];
    const float* outp_b      = (const float*)d_in[12];
    const float* out_W       = (const float*)d_in[13];
    const float* out_b       = (const float*)d_in[14];
    float* out = (float*)d_out;

    char* ws = (char*)d_ws;
    size_t off = 0;
    auto carve = [&](size_t bytes) -> void* {
        void* p = ws + off;
        off += (bytes + 255) & ~(size_t)255;
        return p;
    };
    float*    Hf32    = (float*)   carve((size_t)LL * DD * 4);
    float*    XNf32   = (float*)   carve((size_t)LL * DD * 4);
    float*    U       = (float*)   carve((size_t)LL * SS * 4);
    float*    Sloc    = (float*)   carve((size_t)LL * SS * 4);
    float*    coeff   = (float*)   carve((size_t)NBB * SS * 4);
    float*    Z       = (float*)   carve((size_t)NCHUNK * SS * 4);
    float*    Carry   = (float*)   carve((size_t)NCHUNK * SS * 4);
    _Float16* Hf16    = (_Float16*)carve((size_t)LL * DD * 2);
    _Float16* XNf16   = (_Float16*)carve((size_t)LL * DD * 2);
    _Float16* St16    = (_Float16*)carve((size_t)LL * SS * 2);
    _Float16* Mix16   = (_Float16*)carve((size_t)LL * DD * 2);
    _Float16* W1h     = (_Float16*)carve((size_t)NBB * SS * DD * 2);
    _Float16* W2h     = (_Float16*)carve((size_t)NBB * DD * SS * 2);
    _Float16* W3h     = (_Float16*)carve((size_t)NBB * DD * DD * 2);
    _Float16* OutWh   = (_Float16*)carve((size_t)VV * DD * 2);

    // Weight conversion (cheap, reruns every call for determinism).
    f32_to_f16_kernel<<<512, 256, 0, stream>>>(in_to_state, W1h, NBB * SS * DD);
    f32_to_f16_kernel<<<512, 256, 0, stream>>>(st_to_hid,   W2h, NBB * DD * SS);
    f32_to_f16_kernel<<<512, 256, 0, stream>>>(outp_W,      W3h, NBB * DD * DD);
    f32_to_f16_kernel<<<2048, 256, 0, stream>>>(out_W,      OutWh, VV * DD);

    coeff_kernel<<<(NBB * SS + 255) / 256, 256, 0, stream>>>(a_diag, g_diag, dt,
                                                             coeff, NBB * SS);
    embed_kernel<<<(LL * DD) / 256, 256, 0, stream>>>(tokens, token_embed,
                                                      pos_embed, Hf32);

    const dim3 gemm_block(256);
    for (int i = 0; i < NBB; ++i) {
        ln_kernel<<<LL, 256, 0, stream>>>(Hf32, ln_w + i * DD, ln_b + i * DD,
                                          XNf32, XNf16);
        // U = XN @ in_to_state_i^T   [L,S]
        gemm_wmma_kernel<<<dim3(SS / 64, LL / 128), gemm_block, 0, stream>>>(
            XNf16, W1h + (size_t)i * SS * DD, U, nullptr, nullptr, nullptr,
            nullptr, LL, SS, DD, 0);
        // diagonal scan -> States (fp16)
        scan_pass1_kernel<<<NCHUNK, SS, 0, stream>>>(U, coeff + i * SS, Sloc, Z);
        scan_pass2_kernel<<<1, SS, 0, stream>>>(Z, coeff + i * SS, Carry);
        scan_pass3_kernel<<<NCHUNK, SS, 0, stream>>>(Sloc, Carry, coeff + i * SS,
                                                     St16);
        // MIXED = gelu(States @ state_to_hidden_i^T + direct_i * XN)  [L,D]
        gemm_wmma_kernel<<<dim3(DD / 64, LL / 128), gemm_block, 0, stream>>>(
            St16, W2h + (size_t)i * DD * SS, nullptr, Mix16, nullptr,
            direct + i * DD, XNf32, LL, DD, SS, 1);
        // H += MIXED @ outp_W_i^T + outp_b_i   [L,D]
        gemm_wmma_kernel<<<dim3(DD / 64, LL / 128), gemm_block, 0, stream>>>(
            Mix16, W3h + (size_t)i * DD * DD, Hf32,
            (i == NBB - 1) ? Hf16 : nullptr, outp_b + i * DD, nullptr, nullptr,
            LL, DD, DD, 2);
    }

    // OUT = H @ out_W^T + out_b   [L,V]
    gemm_wmma_kernel<<<dim3(VV / 64, LL / 128), gemm_block, 0, stream>>>(
        Hf16, OutWh, out, nullptr, out_b, nullptr, nullptr, LL, VV, DD, 3);
}
